// GNN_clf_27762668601597
// MI455X (gfx1250) — matrix operand order repaired and checked by reading, not compiled, not run on hardware
//
#include <hip/hip_runtime.h>

// ---------------------------------------------------------------------------
// CDNA5 (gfx1250) GNN pipeline.
//  - All GEMMs: v_wmma_f32_16x16x32_bf16, wave32, 8 waves/block, 128x128 tile.
//  - Double-buffered LDS tiles filled with GLOBAL_LOAD_ASYNC_TO_LDS_B128
//    (ASYNCcnt-tracked), one barrier per k-step.
//  - Weight operands stored transposed [N][K] so B-tiles are contiguous async
//    copies; fragments are pure ds_load_b128 pairs.
// ---------------------------------------------------------------------------

typedef __attribute__((ext_vector_type(16))) __bf16 v16bf;
typedef __attribute__((ext_vector_type(8)))  __bf16 v8bf;
typedef __attribute__((ext_vector_type(8)))  float  v8f;
typedef __attribute__((ext_vector_type(4)))  float  v4f;

#define TILE_M 128
#define TILE_N 128
#define TILE_K 32
#define AT_STRIDE 48   // bf16 elems; 96B rows -> 16B-aligned b128, bank spread
#define BT_STRIDE 48

// ---- CDNA5 async global->LDS copy (16B per lane), ASYNCcnt tracked --------
__device__ __forceinline__ void async_copy_b128(const void* g, void* l) {
#if defined(__AMDGCN__) && __has_builtin(__builtin_amdgcn_global_load_async_to_lds_b128)
  typedef int v4i_ __attribute__((vector_size(16)));
  typedef __attribute__((address_space(1))) v4i_ as1_v4i;  // global
  typedef __attribute__((address_space(3))) v4i_ as3_v4i;  // LDS
  __builtin_amdgcn_global_load_async_to_lds_b128(
      (as1_v4i*)g, (as3_v4i*)l, 0, 0);
#elif defined(__AMDGCN__)
  unsigned loff = (unsigned)(unsigned long long)l;  // low 32 bits = LDS offset
  asm volatile("global_load_async_to_lds_b128 %0, %1, off"
               :: "v"(loff), "v"(g) : "memory");
#else
  (void)g; (void)l;
#endif
}

__device__ __forceinline__ void wait_asynccnt0() {
#if defined(__AMDGCN__) && __has_builtin(__builtin_amdgcn_s_wait_asynccnt)
  __builtin_amdgcn_s_wait_asynccnt(0);
#elif defined(__AMDGCN__)
  asm volatile("s_wait_asynccnt 0x0" ::: "memory");
#endif
}

// Generic fused GEMM:  out = epilogue( in[M,K] @ Wt^T )  with Wt stored [Nc][K].
//   epilogue: (acc + bias[n] + add[m,n]) * scale, optional relu,
//             bf16 store (row-major, or transposed [n][m] when storeT).
// IN_F32: A operand is fp32 (adjacency) -> converted to bf16 while staging.
template <bool IN_F32>
__global__ __launch_bounds__(256)
void gemm_wmma_bf16(const void* __restrict__ inv, long long inBatch, int ldIn,
                    const __bf16* __restrict__ wT, long long wBatch, int ldWT,
                    const float* __restrict__ bias,
                    const __bf16* __restrict__ add, long long addBatch,
                    float scale, int relu, int storeT,
                    __bf16* __restrict__ out, long long outBatch, int ldOut,
                    int M, int K, int Ncols) {
  __shared__ __align__(16) __bf16 aT[2][TILE_M * AT_STRIDE];  // [row][k]
  __shared__ __align__(16) __bf16 bT[2][TILE_N * BT_STRIDE];  // [n][k]

  const int b       = blockIdx.z;
  const int rowBase = blockIdx.y * TILE_M;
  const int nBase   = blockIdx.x * TILE_N;
  const int tid     = threadIdx.x;
  const int lane    = tid & 31;   // wave32
  const int wv      = tid >> 5;   // 8 waves
  const int half    = lane >> 4;
  const int l15     = lane & 15;
  const int wRow    = wv * 16;

  const __bf16* inB = IN_F32 ? nullptr : (const __bf16*)inv + (size_t)b * inBatch;
  const float*  inF = IN_F32 ? (const float*)inv + (size_t)b * inBatch : nullptr;
  const __bf16* wTB = wT + (size_t)b * wBatch;

  // stage one 128x32 A tile into aT[st]
  auto issueA = [&](int k0, int st) {
#pragma unroll
    for (int c = tid; c < 512; c += 256) {
      const int r = c >> 2, k8 = (c & 3) * 8;
      if (IN_F32) {
        const float* src = inF + (size_t)(rowBase + r) * ldIn + k0 + k8;
        v4f f0 = *(const v4f*)src;
        v4f f1 = *(const v4f*)(src + 4);
        v8bf d;
#pragma unroll
        for (int j = 0; j < 4; ++j) { d[j] = (__bf16)f0[j]; d[4 + j] = (__bf16)f1[j]; }
        *(v8bf*)&aT[st][r * AT_STRIDE + k8] = d;
      } else {
        async_copy_b128(inB + (size_t)(rowBase + r) * ldIn + k0 + k8,
                        &aT[st][r * AT_STRIDE + k8]);
      }
    }
  };
  // stage one 128x32 B tile (from transposed weights) into bT[st]
  auto issueB = [&](int k0, int st) {
#pragma unroll
    for (int c = tid; c < 512; c += 256) {
      const int n = c >> 2, k8 = (c & 3) * 8;
      async_copy_b128(wTB + (size_t)(nBase + n) * ldWT + k0 + k8,
                      &bT[st][n * BT_STRIDE + k8]);
    }
  };
  // B fragment: lane = column n; lanes<16 K={0..7,16..23}; lanes>=16 K +8
  auto loadBfrag = [&](int st, int t) {
    const int n = t * 16 + l15;
    v8bf blo = *(const v8bf*)&bT[st][n * BT_STRIDE + half * 8];
    v8bf bhi = *(const v8bf*)&bT[st][n * BT_STRIDE + 16 + half * 8];
    v16bf f;
#pragma unroll
    for (int j = 0; j < 8; ++j) { f[j] = blo[j]; f[8 + j] = bhi[j]; }
    return f;
  };

  v8f acc[8];
#pragma unroll
  for (int t = 0; t < 8; ++t)
#pragma unroll
    for (int r = 0; r < 8; ++r) acc[t][r] = 0.0f;

  issueA(0, 0);
  issueB(0, 0);

  const int KT = K / TILE_K;
  for (int kt = 0; kt < KT; ++kt) {
    const int cur = kt & 1;
    wait_asynccnt0();      // this wave's copies for tile kt complete
    __syncthreads();       // everyone's complete; prior reads of other buffer done
    if (kt + 1 < KT) {     // prefetch next tile into the other buffer
      issueA((kt + 1) * TILE_K, cur ^ 1);
      issueB((kt + 1) * TILE_K, cur ^ 1);
    }

    // A fragment: lanes<16 hold M=l15, K={0..7,16..23}; lanes>=16 K +8
    v8bf alo = *(const v8bf*)&aT[cur][(wRow + l15) * AT_STRIDE + half * 8];
    v8bf ahi = *(const v8bf*)&aT[cur][(wRow + l15) * AT_STRIDE + 16 + half * 8];
    v16bf aF;
#pragma unroll
    for (int j = 0; j < 8; ++j) { aF[j] = alo[j]; aF[8 + j] = ahi[j]; }

    // software-pipelined B fragments so WMMAs don't wait on dscnt==0
    v16bf bF = loadBfrag(cur, 0);
#pragma unroll
    for (int t = 0; t < 8; ++t) {
      v16bf bN;
      if (t < 7) bN = loadBfrag(cur, t + 1);
      acc[t] = __builtin_amdgcn_wmma_f32_16x16x32_bf16(
          false, aF, false, bF, (short)0, acc[t], false, false);
      bF = bN;
    }
    // no trailing barrier: next iteration's wait+barrier provides the fence
  }

  // ---- fused epilogue: C/D layout  M = r + 8*half, N = l15 + 16*t ----
  const __bf16* addB = add ? add + (size_t)b * addBatch : nullptr;
  __bf16* outB = out + (size_t)b * outBatch;
#pragma unroll
  for (int t = 0; t < 8; ++t) {
    const int n = nBase + t * 16 + l15;
    const float bv = bias ? bias[n] : 0.0f;
#pragma unroll
    for (int r = 0; r < 8; ++r) {
      const int m = rowBase + wRow + half * 8 + r;
      float v = acc[t][r] + bv;
      if (addB) v += (float)addB[(size_t)m * Ncols + n];
      v *= scale;
      if (relu && v < 0.0f) v = 0.0f;
      if (storeT) outB[(size_t)n * ldOut + m] = (__bf16)v;
      else        outB[(size_t)m * ldOut + n] = (__bf16)v;
    }
  }
}

__global__ void f32_to_bf16_kernel(const float* __restrict__ src,
                                   __bf16* __restrict__ dst, int n) {
  int i = blockIdx.x * blockDim.x + threadIdx.x;
  if (i < n) dst[i] = (__bf16)src[i];
}

// fp32 [K][Nc] -> bf16 transposed [Nc][K]
__global__ void f32_to_bf16_T_kernel(const float* __restrict__ src,
                                     __bf16* __restrict__ dst, int K, int Nc) {
  int i = blockIdx.x * blockDim.x + threadIdx.x;
  if (i < K * Nc) {
    int k = i / Nc, n = i - k * Nc;
    dst[(size_t)n * K + k] = (__bf16)src[i];
  }
}

// Final layer: 128 -> 1 dot product + sigmoid, fp32 output.
__global__ void out_head_kernel(const __bf16* __restrict__ h,
                                const float* __restrict__ W,
                                const float* __restrict__ b,
                                float* __restrict__ out, int M) {
  int m = blockIdx.x * blockDim.x + threadIdx.x;
  if (m >= M) return;
  const __bf16* row = h + (size_t)m * 128;
  float s = b[0];
#pragma unroll 8
  for (int k = 0; k < 128; ++k) s += (float)row[k] * W[k];
  out[m] = 1.0f / (1.0f + __expf(-s));
}

// ---------------------------------------------------------------------------
extern "C" void kernel_launch(void* const* d_in, const int* in_sizes, int n_in,
                              void* d_out, int out_size, void* d_ws, size_t ws_size,
                              hipStream_t stream) {
  (void)in_sizes; (void)n_in; (void)out_size; (void)ws_size;
  const int B = 8, N = 2048, Mrows = B * N;  // 16384

  const float* X = (const float*)d_in[0];  // [8,2048,64]
  const float* A = (const float*)d_in[1];  // [8,2048,2048]

  // params flattened in dict-insertion order:
  // encoder(4x(W,b)), node(3x4x(W,b)), edge(3x..), node2(3x..), out(4x(W,b))
  struct L { const float* W; const float* b; int din, dout; __bf16* WbfT; };
  static const int encDims[5] = {64, 128, 256, 128, 128};
  static const int midDims[5] = {128, 128, 256, 128, 128};
  static const int outDims[5] = {128, 128, 256, 128, 1};
  L layers[44];
  int p = 2, li = 0;
  auto take4 = [&](const int* dims) {
    for (int i = 0; i < 4; ++i) {
      layers[li].W = (const float*)d_in[p++];
      layers[li].b = (const float*)d_in[p++];
      layers[li].din = dims[i];
      layers[li].dout = dims[i + 1];
      ++li;
    }
  };
  take4(encDims);                                  // 0..3   encoder
  for (int s = 0; s < 3; ++s) take4(midDims);      // 4..15  node
  for (int s = 0; s < 3; ++s) take4(midDims);      // 16..27 edge
  for (int s = 0; s < 3; ++s) take4(midDims);      // 28..39 node2
  take4(outDims);                                  // 40..43 out

  // workspace layout (256B-aligned carve-outs)
  char* wsp = (char*)d_ws;
  size_t off = 0;
  auto walloc = [&](size_t bytes) -> void* {
    void* r = wsp + off;
    off = (off + bytes + 255) & ~(size_t)255;
    return r;
  };
  __bf16* Xbf    = (__bf16*)walloc((size_t)Mrows * 64 * 2);
  __bf16* bufAv  = (__bf16*)walloc((size_t)Mrows * 256 * 2);
  __bf16* bufBv  = (__bf16*)walloc((size_t)Mrows * 256 * 2);
  __bf16* bufH   = (__bf16*)walloc((size_t)Mrows * 128 * 2);
  __bf16* bufHN  = (__bf16*)walloc((size_t)Mrows * 128 * 2);
  __bf16* bufHET = (__bf16*)walloc((size_t)Mrows * 128 * 2);  // he transposed [128][Mrows]
  for (int i = 0; i < 44; ++i)
    layers[i].WbfT = (__bf16*)walloc((size_t)layers[i].din * layers[i].dout * 2);

  // convert inputs/weights to bf16 (weights transposed to [dout][din])
  f32_to_bf16_kernel<<<dim3((Mrows * 64 + 255) / 256), dim3(256), 0, stream>>>(
      X, Xbf, Mrows * 64);
  for (int i = 0; i < 44; ++i) {
    int n = layers[i].din * layers[i].dout;
    f32_to_bf16_T_kernel<<<dim3((n + 255) / 256), dim3(256), 0, stream>>>(
        layers[i].W, layers[i].WbfT, layers[i].din, layers[i].dout);
  }

  auto gemm = [&](const void* in, bool inF32, long long inB, int ldIn,
                  const __bf16* wT, long long wB, int ldWT, const float* bias,
                  const __bf16* add, long long addB, float scale, int relu,
                  int storeT, __bf16* o, long long outB, int ldOut,
                  int M, int K, int Nc, int batches) {
    dim3 g(Nc / TILE_N, M / TILE_M, batches), blk(256);
    if (inF32)
      gemm_wmma_bf16<true><<<g, blk, 0, stream>>>(
          in, inB, ldIn, wT, wB, ldWT, bias, add, addB, scale, relu, storeT,
          o, outB, ldOut, M, K, Nc);
    else
      gemm_wmma_bf16<false><<<g, blk, 0, stream>>>(
          in, inB, ldIn, wT, wB, ldWT, bias, add, addB, scale, relu, storeT,
          o, outB, ldOut, M, K, Nc);
  };

  // 4-layer MLP (ReLU each layer); final layer -> dst with optional transpose
  auto runMLP4 = [&](const __bf16* in0, L* ly, __bf16* dst, int dstT, int dstLd) {
    const __bf16* cur = in0;
    for (int i = 0; i < 4; ++i) {
      __bf16* o = (i == 3) ? dst : ((i & 1) ? bufBv : bufAv);
      int st = (i == 3) ? dstT : 0;
      int ld = (i == 3) ? dstLd : ly[i].dout;
      gemm(cur, false, 0, ly[i].din, ly[i].WbfT, 0, ly[i].din, ly[i].b,
           nullptr, 0, 1.0f, 1, st, o, 0, ld, Mrows, ly[i].din, ly[i].dout, 1);
      cur = o;
    }
  };

  // h = encoder(X)
  runMLP4(Xbf, &layers[0], bufH, 0, 128);

  // 3 message-passing steps
  for (int s = 0; s < 3; ++s) {
    runMLP4(bufH, &layers[4 + 4 * s], bufHN, 0, 128);        // hn
    runMLP4(bufH, &layers[16 + 4 * s], bufHET, 1, Mrows);    // he, stored [n][m]
    // h = (hn + A @ he) / N  — batched; fp32 A staged->bf16, heT async B-tiles
    gemm(A, true, (long long)N * N, N, bufHET, (long long)N, Mrows,
         nullptr, bufHN, (long long)N * 128, 1.0f / (float)N, 0, 0,
         bufH, (long long)N * 128, 128, N, N, 128, B);
    runMLP4(bufH, &layers[28 + 4 * s], bufH, 0, 128);        // node2
  }

  // output stack: 3 ReLU GEMMs + sigmoid head
  L* ol = &layers[40];
  gemm(bufH, false, 0, 128, ol[0].WbfT, 0, 128, ol[0].b, nullptr, 0, 1.0f, 1, 0,
       bufAv, 0, 128, Mrows, 128, 128, 1);
  gemm(bufAv, false, 0, 128, ol[1].WbfT, 0, 128, ol[1].b, nullptr, 0, 1.0f, 1, 0,
       bufBv, 0, 256, Mrows, 128, 256, 1);
  gemm(bufBv, false, 0, 256, ol[2].WbfT, 0, 256, ol[2].b, nullptr, 0, 1.0f, 1, 0,
       bufAv, 0, 128, Mrows, 256, 128, 1);
  out_head_kernel<<<dim3((Mrows + 255) / 256), dim3(256), 0, stream>>>(
      bufAv, ol[3].W, ol[3].b, (float*)d_out, Mrows);
}
